// RNNModel_46162308497610
// MI455X (gfx1250) — matrix operand order repaired and checked by reading, not compiled, not run on hardware
//
#include <hip/hip_runtime.h>
#include <hip/hip_bf16.h>
#include <math.h>

#define V_ 32000
#define E_ 128
#define H_ 128
#define B_ 8
#define T_ 512

#define AS1 __attribute__((address_space(1)))
#define AS3 __attribute__((address_space(3)))

typedef __attribute__((ext_vector_type(16))) __bf16 v16bf;
typedef __attribute__((ext_vector_type(8)))  float  v8f;
typedef unsigned int u32x4 __attribute__((ext_vector_type(4)));
typedef int          i32x4 __attribute__((ext_vector_type(4)));
typedef int          i32x8 __attribute__((ext_vector_type(8)));
typedef int          v4i_g __attribute__((vector_size(16)));   // gcc-style, matches builtin proto

union Frag {
    v16bf  v;
    uint4  q[2];
    __bf16 h[16];
};

#if __has_builtin(__builtin_amdgcn_global_load_async_to_lds_b128) && \
    __has_builtin(__builtin_amdgcn_s_wait_asynccnt)
#define USE_ASYNC 1
#else
#define USE_ASYNC 0
#endif

#if __has_builtin(__builtin_amdgcn_tensor_load_to_lds) && \
    __has_builtin(__builtin_amdgcn_s_wait_tensorcnt)
#define USE_TDM 1
#else
#define USE_TDM 0
#endif

__device__ __forceinline__ float fast_tanh(float x) {
#if __has_builtin(__builtin_amdgcn_tanhf)
    return __builtin_amdgcn_tanhf(x);   // v_tanh_f32 (TRANS32)
#else
    return tanhf(x);
#endif
}

#if USE_ASYNC
__device__ __forceinline__ void async_b128(const void* g, void* l) {
    __builtin_amdgcn_global_load_async_to_lds_b128(
        (AS1 v4i_g*)(void*)g, (AS3 v4i_g*)l, 0, 0);
}
#endif

// ---------------------------------------------------------------------------
// Kernel 1: fc_W f32 -> bf16 (one-time; 8 MB, stays L2-resident for big GEMM)
// ---------------------------------------------------------------------------
__global__ __launch_bounds__(256)
void k_cvt_fcw(const float* __restrict__ fcW, unsigned short* __restrict__ outw, int n) {
    int i = blockIdx.x * blockDim.x + threadIdx.x;
    if (i < n) {
        __bf16 b = (__bf16)fcW[i];
        outw[i] = __builtin_bit_cast(unsigned short, b);
    }
}

// ---------------------------------------------------------------------------
// Kernel 2: pre0[t][brow][j] = emb[x[brow,t]] . W_ih0[j] + b_ih0[j] + b_hh0[j]
// Layout [T][16][H] f32, rows 8..15 zero (M-padding for WMMA).
// ---------------------------------------------------------------------------
__global__ __launch_bounds__(128)
void k_pre0(const int* __restrict__ x, const float* __restrict__ emb,
            const float* __restrict__ Wih0, const float* __restrict__ bih0,
            const float* __restrict__ bhh0, float* __restrict__ pre0) {
    int j  = threadIdx.x;          // 0..127
    int t  = blockIdx.x >> 4;
    int br = blockIdx.x & 15;
    float* outp = pre0 + (size_t)blockIdx.x * H_ + j;
    if (br >= B_) { *outp = 0.f; return; }
    int tok = x[br * T_ + t];      // x is [B,T]
    const float4* e4 = (const float4*)(emb  + (size_t)tok * E_);
    const float4* w4 = (const float4*)(Wih0 + (size_t)j   * E_);
    float acc = 0.f;
#pragma unroll
    for (int k = 0; k < E_ / 4; ++k) {
        float4 a = e4[k], b = w4[k];
        acc += a.x * b.x + a.y * b.y + a.z * b.z + a.w * b.w;
    }
    *outp = acc + bih0[j] + bhh0[j];
}

// ---------------------------------------------------------------------------
// Kernel 3: fused two-layer RNN scan. One workgroup, 8 waves.
//   phase1: h0 = tanh(pre0_t + h0 W_hh0^T)                (K=128, 4 WMMA/wave)
//   phase2: h1 = tanh([h0_t;h1] [W_ih1;W_hh1]^T + bias)   (K=256, 8 WMMA/wave)
// Weight B-fragments in registers; h0/h1 in LDS; pre0 double-buffered into
// LDS via async global->LDS loads (one step ahead of the recurrence).
// ---------------------------------------------------------------------------
__global__ __launch_bounds__(256)
void k_scan(const float* __restrict__ pre0,
            const float* __restrict__ Whh0,
            const float* __restrict__ Wih1, const float* __restrict__ Whh1,
            const float* __restrict__ bih1, const float* __restrict__ bhh1,
            unsigned short* __restrict__ h1_out) {
    __shared__ __bf16 lds_h0[16 * H_];
    __shared__ __bf16 lds_h1[16 * H_];
#if USE_ASYNC
    __shared__ float lds_pre[2][16 * H_];
#endif

    const int tid   = threadIdx.x;
    const int wv    = tid >> 5;        // 0..7 -> N-tile
    const int lane  = tid & 31;
    const int r     = lane & 15;       // A row / C column index
    const int hi    = lane >> 4;       // half-wave select
    const int abase = hi * 16;         // A frag K sub-base
    const int bbase = hi * 16;         // B frag K sub-base
    const int crow0 = hi * 8;          // C/D row base
    const int n     = wv * 16 + r;     // output column owned by this lane

    for (int i = tid; i < 16 * H_; i += 256) {
        lds_h0[i] = (__bf16)0.f;
        lds_h1[i] = (__bf16)0.f;
    }

#if USE_ASYNC
    {   // prologue: stage pre0[t=0] into buffer 0 (32 B per thread)
        const float* g = pre0 + tid * 8;
        async_b128(g,     &lds_pre[0][tid * 8]);
        async_b128(g + 4, &lds_pre[0][tid * 8 + 4]);
    }
#endif

    // Preload B fragments (columns of B == rows of the [N,K] weight matrices).
    Frag bw0[4];
#pragma unroll
    for (int kb = 0; kb < 4; ++kb) {
        const float* row = Whh0 + (size_t)n * H_ + kb * 32 + bbase;
#pragma unroll
        for (int i = 0; i < 16; ++i) bw0[kb].h[i] = (__bf16)row[i];
    }
    Frag bw1[8];
#pragma unroll
    for (int kb = 0; kb < 8; ++kb) {
        const float* row = (kb < 4)
            ? (Wih1 + (size_t)n * H_ + kb * 32 + bbase)
            : (Whh1 + (size_t)n * H_ + (kb - 4) * 32 + bbase);
#pragma unroll
        for (int i = 0; i < 16; ++i) bw1[kb].h[i] = (__bf16)row[i];
    }
    const float bias1 = bih1[n] + bhh1[n];
    __syncthreads();

    for (int t = 0; t < T_; ++t) {
#if USE_ASYNC
        // Prefetch next step's pre0 tile; wait only for the previous batch.
        if (t + 1 < T_) {
            const float* g = pre0 + (size_t)(t + 1) * (16 * H_) + tid * 8;
            float*       l = &lds_pre[(t + 1) & 1][tid * 8];
            async_b128(g,     l);
            async_b128(g + 4, l + 4);
            __builtin_amdgcn_s_wait_asynccnt(2);
        } else {
            __builtin_amdgcn_s_wait_asynccnt(0);
        }
        __syncthreads();
        const float* pt = lds_pre[t & 1];
#else
        const float* pt = pre0 + (size_t)t * 16 * H_;
#endif
        // ---------------- layer 0 recurrence ----------------
        v8f acc;
#pragma unroll
        for (int vv = 0; vv < 8; ++vv) acc[vv] = pt[(crow0 + vv) * H_ + n];
#pragma unroll
        for (int kb = 0; kb < 4; ++kb) {
            Frag a;
            a.q[0] = *(const uint4*)(&lds_h0[r * H_ + kb * 32 + abase]);
            a.q[1] = *(const uint4*)(&lds_h0[r * H_ + kb * 32 + abase + 8]);
            acc = __builtin_amdgcn_wmma_f32_16x16x32_bf16(
                false, a.v, false, bw0[kb].v, (short)0, acc, false, false);
        }
        float h0n[8];
#pragma unroll
        for (int vv = 0; vv < 8; ++vv) h0n[vv] = fast_tanh(acc[vv]);
        __syncthreads();
#pragma unroll
        for (int vv = 0; vv < 8; ++vv) lds_h0[(crow0 + vv) * H_ + n] = (__bf16)h0n[vv];
        __syncthreads();

        // ------- layer 1 (input GEMM + recurrence fused, K=256) -------
        v8f acc2;
#pragma unroll
        for (int vv = 0; vv < 8; ++vv) acc2[vv] = bias1;
#pragma unroll
        for (int kb = 0; kb < 8; ++kb) {
            const __bf16* src = (kb < 4) ? lds_h0 : lds_h1;
            const int kk = (kb & 3) * 32 + abase;
            Frag a;
            a.q[0] = *(const uint4*)(&src[r * H_ + kk]);
            a.q[1] = *(const uint4*)(&src[r * H_ + kk + 8]);
            acc2 = __builtin_amdgcn_wmma_f32_16x16x32_bf16(
                false, a.v, false, bw1[kb].v, (short)0, acc2, false, false);
        }
        float h1n[8];
#pragma unroll
        for (int vv = 0; vv < 8; ++vv) h1n[vv] = fast_tanh(acc2[vv]);
        __syncthreads();
#pragma unroll
        for (int vv = 0; vv < 8; ++vv) lds_h1[(crow0 + vv) * H_ + n] = (__bf16)h1n[vv];
        if (hi == 0) {  // rows 0..7 are the real batch rows
#pragma unroll
            for (int vv = 0; vv < 8; ++vv) {
                __bf16 b = (__bf16)h1n[vv];
                h1_out[((size_t)t * B_ + vv) * H_ + n] =
                    __builtin_bit_cast(unsigned short, b);
            }
        }
        __syncthreads();
    }
}

// ---------------------------------------------------------------------------
// Kernel 4: logits = h1[4096x128] * fc_W^T[128x32000] + fc_b  (store-bound)
// Block: 8 waves; 1 M-tile x 32 N-tiles. A tile (4 KB) staged into LDS via
// TDM (tensor_load_to_lds) when available, else async-LDS, else plain loads.
// Output layout btv: m = t*8+b  ->  out[(b*T + t)*V + n].
// ---------------------------------------------------------------------------
__global__ __launch_bounds__(256)
void k_logits(const unsigned short* __restrict__ h1w,
              const unsigned short* __restrict__ fcWw,
              const float* __restrict__ fcb, float* __restrict__ out) {
    const __bf16* fcWb = (const __bf16*)fcWw;
    __shared__ __bf16 lds_a[16 * H_];   // 4 KB A tile

    const int tid   = threadIdx.x;
    const int wv    = tid >> 5, lane = tid & 31;
    const int r     = lane & 15, hi = lane >> 4;
    const int abase = hi * 16, bbase = hi * 16, crow0 = hi * 8;
    const int mt    = blockIdx.y;
    const int nt0   = blockIdx.x * 32 + wv * 4;

    const size_t a_elems = (size_t)mt * 16 * H_;   // element offset of A tile

#if USE_TDM
    if (wv == 0) {
        unsigned long long ga =
            (unsigned long long)(uintptr_t)(h1w + a_elems);
        unsigned lds_off = (unsigned)(uintptr_t)(AS3 char*)&lds_a[0];
        // D# group0: count=1 | lds_addr | global_addr | type=2
        u32x4 g0;
        g0[0] = 1u;
        g0[1] = lds_off;
        g0[2] = (unsigned)(ga & 0xFFFFFFFFu);
        g0[3] = (unsigned)((ga >> 32) & 0x01FFFFFFu) | (2u << 30);
        // D# group1: data_size=2B, tensor_dim0=2048, tensor_dim1=1,
        //            tile_dim0=2048 (one contiguous 4KB row), tile_dim1=1
        i32x8 g1;
        g1[0] = 0x00010000;                       // wg_mask=0, data_size=1 (2B)
        g1[1] = (int)((2048u & 0xFFFFu) << 16);   // tensor_dim0[15:0]
        g1[2] = (int)(1u << 16);                  // dim0[31:16]=0 | dim1[15:0]=1
        g1[3] = (int)(2048u << 16);               // dim1[31:16]=0 | tile_dim0=2048
        g1[4] = 1;                                // tile_dim1=1, tile_dim2=0
        g1[5] = 2048;                             // tensor_dim0_stride[31:0]
        g1[6] = (int)(2048u << 16);               // stride0[47:32]=0|stride1[15:0]
        g1[7] = 0;                                // stride1[47:16]
        i32x4 gz = {0, 0, 0, 0};
#if __clang_major__ >= 23
        i32x8 gz8 = {0, 0, 0, 0, 0, 0, 0, 0};
        __builtin_amdgcn_tensor_load_to_lds(g0, g1, gz, gz, gz8, 0);
#else
        __builtin_amdgcn_tensor_load_to_lds(g0, g1, gz, gz, 0);
#endif
        __builtin_amdgcn_s_wait_tensorcnt(0);
    }
    __syncthreads();
#elif USE_ASYNC
    async_b128((const char*)h1w + a_elems * 2 + tid * 16,
               (char*)lds_a + tid * 16);
    __builtin_amdgcn_s_wait_asynccnt(0);
    __syncthreads();
#else
    ((uint4*)lds_a)[tid] = ((const uint4*)(h1w + a_elems))[tid];
    __syncthreads();
#endif

    // A fragments for this M tile (reused by all 4 N tiles of this wave)
    Frag afr[4];
#pragma unroll
    for (int kb = 0; kb < 4; ++kb) {
        afr[kb].q[0] = *(const uint4*)(&lds_a[r * H_ + kb * 32 + abase]);
        afr[kb].q[1] = *(const uint4*)(&lds_a[r * H_ + kb * 32 + abase + 8]);
    }

#pragma unroll
    for (int j = 0; j < 4; ++j) {
        const int nt = nt0 + j;
        if (nt < V_ / 16) {                 // wave-uniform guard
            const int n = nt * 16 + r;
            const __bf16* brow = fcWb + (size_t)n * H_;
            if (j < 3) __builtin_prefetch(fcWb + (size_t)(n + 16) * H_, 0, 1);
            const float bias = fcb[n];
            v8f acc;
#pragma unroll
            for (int vv = 0; vv < 8; ++vv) acc[vv] = bias;
#pragma unroll
            for (int kb = 0; kb < 4; ++kb) {
                Frag b;
                b.q[0] = *(const uint4*)(brow + kb * 32 + bbase);
                b.q[1] = *(const uint4*)(brow + kb * 32 + bbase + 8);
                acc = __builtin_amdgcn_wmma_f32_16x16x32_bf16(
                    false, afr[kb].v, false, b.v, (short)0, acc, false, false);
            }
#pragma unroll
            for (int vv = 0; vv < 8; ++vv) {
                const int m  = mt * 16 + crow0 + vv;
                const int b_ = m & 7;
                const int t  = m >> 3;
                out[((size_t)b_ * T_ + t) * (size_t)V_ + n] = acc[vv];
            }
        }
    }
}

// ---------------------------------------------------------------------------
extern "C" void kernel_launch(void* const* d_in, const int* in_sizes, int n_in,
                              void* d_out, int out_size, void* d_ws, size_t ws_size,
                              hipStream_t stream) {
    const int*   x    = (const int*)  d_in[0];
    const float* emb  = (const float*)d_in[1];
    const float* Wih0 = (const float*)d_in[2];
    const float* Whh0 = (const float*)d_in[3];
    const float* bih0 = (const float*)d_in[4];
    const float* bhh0 = (const float*)d_in[5];
    const float* Wih1 = (const float*)d_in[6];
    const float* Whh1 = (const float*)d_in[7];
    const float* bih1 = (const float*)d_in[8];
    const float* bhh1 = (const float*)d_in[9];
    const float* fcW  = (const float*)d_in[10];
    const float* fcb  = (const float*)d_in[11];
    float* out = (float*)d_out;

    // Workspace: fcW_bf16 (8,192,000 B) | pre0 (4,194,304 B) | h1_bf16 (1,048,576 B)
    char* ws = (char*)d_ws;
    unsigned short* fcWb = (unsigned short*)ws;
    float*          pre0 = (float*)(ws + 8192000);
    unsigned short* h1b  = (unsigned short*)(ws + 8192000 + 4194304);

    k_cvt_fcw<<<(V_ * H_ + 255) / 256, 256, 0, stream>>>(fcW, fcWb, V_ * H_);
    k_pre0<<<T_ * 16, 128, 0, stream>>>(x, emb, Wih0, bih0, bhh0, pre0);
    k_scan<<<1, 256, 0, stream>>>(pre0, Whh0, Wih1, Whh1, bih1, bhh1, h1b);
    k_logits<<<dim3(63, 256), 256, 0, stream>>>(h1b, fcWb, fcb, out);
}
